// MultiHeadAttention_71768903516761
// MI455X (gfx1250) — matrix operand order repaired and checked by reading, not compiled, not run on hardware
//
#include <hip/hip_runtime.h>
#include <math.h>

// ---------------------------------------------------------------------------
// MHA forward for MI455X (gfx1250), wave32, WMMA f32_16x16x32_f16,
// async GLOBAL->LDS staging (ASYNCcnt) with double buffering.
// B=2, S=2048, D=1024, H=16, HD=64.
// d_out = [out (B*S*D f32)] ++ [attn (B*H*S*S f32)]
// ---------------------------------------------------------------------------

#define B_   2
#define S_   2048
#define D_   1024
#define H_   16
#define HD_  64

typedef __attribute__((ext_vector_type(16))) _Float16 v16h;
typedef __attribute__((ext_vector_type(8)))  float    v8f;

union FragH {
    v16h      v;
    _Float16  h[16];
    unsigned  u[8];
};

// A-matrix (16x32 f16) per-lane K index for vgpr-half v (0..7), lane-half hi (0/1).
// Contiguous order: lanes 0-15 hold K 0..15, lanes 16-31 hold K 16..31 (matches B).
__device__ __forceinline__ int khalf_idx(int v, int hi) {
    return 2 * v + hi * 16;
}

// Async copy 16B global -> LDS per lane (ASYNCcnt tracked). LDS offset is the
// low 32 bits of the flat pointer (ISA 10.2: LDS aperture truncates to addr[31:0]).
__device__ __forceinline__ void async_b128(void* lds_dst, const void* gsrc) {
    unsigned loff = (unsigned)(uintptr_t)lds_dst;
    unsigned long long ga = (unsigned long long)(uintptr_t)gsrc;
    asm volatile("global_load_async_to_lds_b128 %0, %1, off"
                 :: "v"(loff), "v"(ga) : "memory");
}
__device__ __forceinline__ void wait_async0() {
    asm volatile("s_wait_asynccnt 0x0" ::: "memory");
}

// ---------------------------------------------------------------------------
// f32 -> f16 conversion (one-shot, removes per-tile redundant cvt in GEMMs)
// ---------------------------------------------------------------------------
__global__ __launch_bounds__(256) void cvt_f16(const float* __restrict__ in,
                                               void* __restrict__ outp, int n)
{
    _Float16* out = (_Float16*)outp;
    const int i = (blockIdx.x * 256 + threadIdx.x) * 8;
    if (i < n) {
#pragma unroll
        for (int j = 0; j < 8; ++j) out[i + j] = (_Float16)in[i + j];
    }
}

// ---------------------------------------------------------------------------
// Tiled GEMM (all-f16 operands):  C[M=4096,N=1024] = A[M,K] * W[N,K]^T
// MODE 0: store f32 to Out[m*N + n]                        (final output)
// MODE 1: store f16 head-major  Out[((b*H+h)*S+s)*HD+hd]   (Q, K)
// MODE 2: store f16 transposed  Out[((b*H+h)*HD+hd)*S+s]   (V^T)
// 256 threads / 8 waves, tile 64(M) x 128(N), k-step 32, double-buffered
// async GLOBAL->LDS staging.
// ---------------------------------------------------------------------------
template <int MODE>
__global__ __launch_bounds__(256) void gemm16(const void* __restrict__ Aptr,
                                              const void* __restrict__ Wptr,
                                              void* __restrict__ Out)
{
    constexpr int Kdim = D_;
    constexpr int Ndim = D_;
    constexpr int LDT  = 40;          // halves: rows 80B (16B-aligned for B128 async,
                                      // bank stride 20 -> conflict-free column walks)
    constexpr int NK   = Kdim / 32;

    __shared__ _Float16 Asl[2][64 * LDT];
    __shared__ _Float16 Bsl[2][128 * LDT];

    const _Float16* A = (const _Float16*)Aptr;
    const _Float16* W = (const _Float16*)Wptr;

    const int tid  = threadIdx.x;
    const int lane = tid & 31;
    const int wid  = tid >> 5;
    const int wm   = wid & 3;         // 4 M-subtiles of 16
    const int wn   = wid >> 2;        // 2 N-halves of 64
    const int m0   = blockIdx.x * 64;
    const int n0   = blockIdx.y * 128;

    const int sr = tid >> 2;          // staging row 0..63
    const int sc8 = (tid & 3) * 8;    // staging col (halves), 16B chunks

    v8f acc[4] = {};

    // issue one k-tile of async copies into buffer `buf`
    auto stage = [&](int buf, int kt) {
        const int k0 = kt * 32;
        async_b128(&Asl[buf][sr * LDT + sc8],
                   A + (size_t)(m0 + sr) * Kdim + k0 + sc8);
        async_b128(&Bsl[buf][sr * LDT + sc8],
                   W + (size_t)(n0 + sr) * Kdim + k0 + sc8);
        async_b128(&Bsl[buf][(sr + 64) * LDT + sc8],
                   W + (size_t)(n0 + sr + 64) * Kdim + k0 + sc8);
    };

    stage(0, 0);

    for (int kt = 0; kt < NK; ++kt) {
        const int p = kt & 1;
        wait_async0();                // my async writes landed in LDS
        __syncthreads();              // everyone's landed; prev-iter reads done
        if (kt + 1 < NK) stage(p ^ 1, kt + 1);  // prefetch next tile

        // ---- A fragment (16x32), contiguous k layout, dword LDS reads ----
        FragH af;
        {
            const int row = lane & 15, hi = lane >> 4;
#pragma unroll
            for (int v2 = 0; v2 < 8; ++v2) {
                const int kk = khalf_idx(v2, hi);
                af.u[v2] = *(const unsigned*)&Asl[p][(wm * 16 + row) * LDT + kk];
            }
        }
        // ---- 4 B fragments + WMMA ----
#pragma unroll
        for (int t = 0; t < 4; ++t) {
            FragH bf;
            const int n  = wn * 64 + t * 16 + (lane & 15);
            const int kb = (lane >> 4) * 16;  // lanes 0-15: K 0..15, 16-31: K 16..31
#pragma unroll
            for (int v2 = 0; v2 < 8; ++v2)
                bf.u[v2] = *(const unsigned*)&Bsl[p][n * LDT + kb + 2 * v2];
            acc[t] = __builtin_amdgcn_wmma_f32_16x16x32_f16(
                false, af.v, false, bf.v, (short)0, acc[t], false, false);
        }
    }

    // ---- store ----
    const int row8 = lane >> 4;       // C layout: lanes 0-15 -> M 0..7, 16-31 -> M 8..15
    const int ncol = lane & 15;
#pragma unroll
    for (int t = 0; t < 4; ++t) {
#pragma unroll
        for (int r = 0; r < 8; ++r) {
            const int gm = m0 + wm * 16 + row8 * 8 + r;
            const int gn = n0 + wn * 64 + t * 16 + ncol;
            const float val = acc[t][r];
            if (MODE == 0) {
                ((float*)Out)[(size_t)gm * Ndim + gn] = val;
            } else {
                const int b = gm >> 11, s = gm & (S_ - 1);
                const int h = gn >> 6,  hd = gn & (HD_ - 1);
                _Float16* O = (_Float16*)Out;
                if (MODE == 1)
                    O[((size_t)(b * H_ + h) * S_ + s) * HD_ + hd] = (_Float16)val;
                else
                    O[((size_t)(b * H_ + h) * HD_ + hd) * S_ + s] = (_Float16)val;
            }
        }
    }
}

// ---------------------------------------------------------------------------
// Fused attention: one block per (b, h, 16-row q tile). 256 threads / 8 waves.
// Phase A: scores 16x2048 via WMMA -> LDS (f32)
// Phase B: softmax in LDS, single NT write of attn to HBM
// Phase C: ctx = P(16x2048) @ V(2048x64) via WMMA from LDS + V^T, f16 ctx out
// ---------------------------------------------------------------------------
__global__ __launch_bounds__(256) void attn_kernel(const void* __restrict__ Qp,
                                                   const void* __restrict__ Kp,
                                                   const void* __restrict__ Vtp,
                                                   float* __restrict__ attn_out,
                                                   void* __restrict__ ctx_outp)
{
    constexpr int LDSS = S_ + 1;   // 2049: odd stride -> conflict-free column walks
    __shared__ float sc[16 * LDSS];
    __shared__ float red[16][16];
    __shared__ float rowmax[16];
    __shared__ float rowsum[16];

    const _Float16* Qh = (const _Float16*)Qp;    // [B,H,S,HD]
    const _Float16* Kh = (const _Float16*)Kp;    // [B,H,S,HD]
    const _Float16* Vt = (const _Float16*)Vtp;   // [B,H,HD,S]
    _Float16* ctx_out  = (_Float16*)ctx_outp;    // [B,S,D] combined heads

    const int tid  = threadIdx.x;
    const int lane = tid & 31;
    const int wid  = tid >> 5;
    const int qt   = blockIdx.x;   // 0..127
    const int h    = blockIdx.y;
    const int b    = blockIdx.z;

    const size_t head = (size_t)b * H_ + h;
    const _Float16* Qb = Qh + (head * S_ + (size_t)qt * 16) * HD_;
    const _Float16* Kb = Kh + head * S_ * HD_;
    const _Float16* Vb = Vt + head * HD_ * S_;

    // ---- Q fragments: 16x64 = two 16x32 A-frags ----
    FragH qa[2];
    {
        const int row = lane & 15, hi = lane >> 4;
#pragma unroll
        for (int half = 0; half < 2; ++half) {
#pragma unroll
            for (int v2 = 0; v2 < 8; ++v2) {
                const int kk = half * 32 + khalf_idx(v2, hi);
                qa[half].h[2 * v2]     = Qb[row * HD_ + kk];
                qa[half].h[2 * v2 + 1] = Qb[row * HD_ + kk + 1];
            }
        }
    }

    // ---- Phase A: scores = (Q K^T) * 1/sqrt(HD) into LDS ----
    for (int kt = wid; kt < S_ / 16; kt += 8) {
        v8f c = {};
#pragma unroll
        for (int half = 0; half < 2; ++half) {
            FragH bf;
            const int n  = kt * 16 + (lane & 15);         // key row
            const int kb = half * 32 + (lane >> 4) * 16;  // hd offset, 32B aligned
            bf.v = *(const v16h*)(Kb + (size_t)n * HD_ + kb);
            c = __builtin_amdgcn_wmma_f32_16x16x32_f16(
                false, qa[half].v, false, bf.v, (short)0, c, false, false);
        }
        const int row8 = lane >> 4, ncol = lane & 15;
#pragma unroll
        for (int r = 0; r < 8; ++r)
            sc[(row8 * 8 + r) * LDSS + kt * 16 + ncol] = c[r] * 0.125f;  // 1/sqrt(64)
    }
    __syncthreads();

    // ---- Phase B: softmax over each of 16 rows (16 threads per row) ----
    const int row = tid >> 4, sub = tid & 15;
    {
        float m = -3.4e38f;
        for (int j = sub; j < S_; j += 16) m = fmaxf(m, sc[row * LDSS + j]);
        red[row][sub] = m;
    }
    __syncthreads();
    if (sub == 0) {
        float m = red[row][0];
#pragma unroll
        for (int j = 1; j < 16; ++j) m = fmaxf(m, red[row][j]);
        rowmax[row] = m;
    }
    __syncthreads();
    {
        const float m = rowmax[row];
        float ssum = 0.f;
        for (int j = sub; j < S_; j += 16) {
            const float p = __expf(sc[row * LDSS + j] - m);
            sc[row * LDSS + j] = p;
            ssum += p;
        }
        red[row][sub] = ssum;
    }
    __syncthreads();
    if (sub == 0) {
        float ssum = 0.f;
#pragma unroll
        for (int j = 0; j < 16; ++j) ssum += red[row][j];
        rowsum[row] = ssum;
    }
    __syncthreads();
    {
        const float inv = 1.0f / rowsum[row];
        float* arow = attn_out + (head * S_ + (size_t)qt * 16 + row) * S_;
        for (int j = sub; j < S_; j += 16) {
            const float p = sc[row * LDSS + j] * inv;
            sc[row * LDSS + j] = p;
            __builtin_nontemporal_store(p, &arow[j]);  // 512MB stream: don't pollute caches
        }
    }
    __syncthreads();

    // ---- Phase C: ctx = P @ V via WMMA (waves 0..3, one 16-col hd tile each) ----
    if (wid < 4) {
        const int ntile = wid;
        v8f c = {};
        const int rowA = lane & 15, hi = lane >> 4;
        const int hd   = ntile * 16 + (lane & 15);
        for (int ks = 0; ks < S_ / 32; ++ks) {
            FragH af;
#pragma unroll
            for (int v2 = 0; v2 < 8; ++v2) {
                const int kk = ks * 32 + khalf_idx(v2, hi);
                af.h[2 * v2]     = (_Float16)sc[rowA * LDSS + kk];
                af.h[2 * v2 + 1] = (_Float16)sc[rowA * LDSS + kk + 1];
            }
            FragH bf;
            const int k16 = ks * 32 + (lane >> 4) * 16;   // 32B aligned in V^T row
            bf.v = *(const v16h*)(Vb + (size_t)hd * S_ + k16);
            c = __builtin_amdgcn_wmma_f32_16x16x32_f16(
                false, af.v, false, bf.v, (short)0, c, false, false);
        }
        const int row8 = lane >> 4, ncol = lane & 15;
#pragma unroll
        for (int r = 0; r < 8; ++r) {
            const int q = qt * 16 + row8 * 8 + r;
            const int e = h * HD_ + ntile * 16 + ncol;
            ctx_out[((size_t)b * S_ + q) * D_ + e] = (_Float16)c[r];
        }
    }
}

// ---------------------------------------------------------------------------
extern "C" void kernel_launch(void* const* d_in, const int* in_sizes, int n_in,
                              void* d_out, int out_size, void* d_ws, size_t ws_size,
                              hipStream_t stream)
{
    (void)in_sizes; (void)n_in; (void)out_size; (void)ws_size;

    const float* q   = (const float*)d_in[0];
    const float* k   = (const float*)d_in[1];
    const float* v   = (const float*)d_in[2];
    const float* w_q = (const float*)d_in[3];
    const float* w_k = (const float*)d_in[4];
    const float* w_v = (const float*)d_in[5];
    const float* w_o = (const float*)d_in[6];

    float* out  = (float*)d_out;                       // B*S*D
    float* attn = out + (size_t)B_ * S_ * D_;          // B*H*S*S

    // workspace layout (f16 segments)
    char* ws = (char*)d_ws;
    const size_t MB = 1u << 20;
    void* qf16  = ws;               // 8 MB  (B*S*D halves)
    void* kf16  = ws + 8 * MB;      // 8 MB
    void* vf16  = ws + 16 * MB;     // 8 MB
    void* wq16  = ws + 24 * MB;     // 2 MB  (D*D halves)
    void* wk16  = ws + 26 * MB;     // 2 MB
    void* wv16  = ws + 28 * MB;     // 2 MB
    void* wo16  = ws + 30 * MB;     // 2 MB
    void* wsQ   = ws + 32 * MB;     // 8 MB  f16 [B,H,S,HD]
    void* wsK   = ws + 40 * MB;     // 8 MB  f16 [B,H,S,HD]
    void* wsVt  = ws + 48 * MB;     // 8 MB  f16 [B,H,HD,S]
    void* wsCtx = ws + 56 * MB;     // 8 MB  f16 [B,S,D]

    const int nAct = B_ * S_ * D_;  // 4M
    const int nWgt = D_ * D_;       // 1M
    cvt_f16<<<nAct / 2048, 256, 0, stream>>>(q, qf16, nAct);
    cvt_f16<<<nAct / 2048, 256, 0, stream>>>(k, kf16, nAct);
    cvt_f16<<<nAct / 2048, 256, 0, stream>>>(v, vf16, nAct);
    cvt_f16<<<nWgt / 2048, 256, 0, stream>>>(w_q, wq16, nWgt);
    cvt_f16<<<nWgt / 2048, 256, 0, stream>>>(w_k, wk16, nWgt);
    cvt_f16<<<nWgt / 2048, 256, 0, stream>>>(w_v, wv16, nWgt);
    cvt_f16<<<nWgt / 2048, 256, 0, stream>>>(w_o, wo16, nWgt);

    const dim3 ggrid((B_ * S_) / 64, D_ / 128);   // (64, 8)
    gemm16<1><<<ggrid, 256, 0, stream>>>(qf16, wq16, wsQ);   // Q f16 [B,H,S,HD]
    gemm16<1><<<ggrid, 256, 0, stream>>>(kf16, wk16, wsK);   // K f16 [B,H,S,HD]
    gemm16<2><<<ggrid, 256, 0, stream>>>(vf16, wv16, wsVt);  // V^T f16 [B,H,HD,S]

    attn_kernel<<<dim3(S_ / 16, H_, B_), 256, 0, stream>>>(wsQ, wsK, wsVt, attn, wsCtx);

    gemm16<0><<<ggrid, 256, 0, stream>>>(wsCtx, wo16, out);  // out f32 [B,S,D]
}
